// HomeExpertSystem_68513318306064
// MI455X (gfx1250) — compile-verified, matrix operand order checked by reading, not run on hardware
//
#include <hip/hip_runtime.h>
#include <hip/hip_bf16.h>
#include <stdint.h>

#define N_CAND     262144
#define TRAJ_LEN   60
#define AREA_SIZE  224
#define AREA_HALF  112
#define N_SELECT   6
#define PENALTY    0.04f
#define MASK_WORDS 1568   // 224*224/32
#define MASK_B128  392    // 1568/4

typedef __attribute__((ext_vector_type(8))) int v8i;
typedef __attribute__((ext_vector_type(4))) int v4i;
typedef __attribute__((address_space(1))) v4i g_v4i;   // global AS pointer target
typedef __attribute__((address_space(3))) v4i l_v4i;   // LDS AS pointer target

// ---------------- Kernel 0: pack da_area into a bitmask (bit = off_da) -------
__global__ __launch_bounds__(256) void pack_kernel(const float* __restrict__ da,
                                                   unsigned* __restrict__ maskw) {
  int w = blockIdx.x * 256 + threadIdx.x;
  if (w < MASK_WORDS) {
    unsigned word = 0u;
#pragma unroll 8
    for (int b = 0; b < 32; ++b)
      word |= (da[w * 32 + b] == 0.0f ? 1u : 0u) << b;
    maskw[w] = word;
  }
}

// ---------------- Kernel 1: score candidates (WMMA IU8 row-sum) --------------
__global__ __launch_bounds__(256) void score_kernel(const float2* __restrict__ traj,
                                                    const float* __restrict__ conf,
                                                    const unsigned* __restrict__ maskg,
                                                    float* __restrict__ scores) {
  __shared__ unsigned smask[MASK_WORDS];
  const int tid = threadIdx.x;

  // ---- stage 6272B bitmask global -> LDS via CDNA5 async path ----
#if __has_builtin(__builtin_amdgcn_global_load_async_to_lds_b128)
  for (int w4 = tid; w4 < MASK_B128; w4 += 256) {
    __builtin_amdgcn_global_load_async_to_lds_b128(
        (g_v4i*)(maskg + 4 * w4),
        (l_v4i*)(&smask[4 * w4]), 0, 0);
  }
#if __has_builtin(__builtin_amdgcn_s_wait_asynccnt)
  __builtin_amdgcn_s_wait_asynccnt(0);
#else
  asm volatile("s_wait_asynccnt 0" ::: "memory");
#endif
#else
  for (int w4 = tid; w4 < MASK_B128; w4 += 256) {
    unsigned lofs = (unsigned)(uintptr_t)(&smask[4 * w4]);
    const unsigned* gp = maskg + 4 * w4;
    asm volatile("global_load_async_to_lds_b128 %0, %1, off"
                 :: "v"(lofs), "v"(gp) : "memory");
  }
  asm volatile("s_wait_asynccnt 0" ::: "memory");
#endif
  __syncthreads();

  const int lane = tid & 31;
  const int wid  = tid >> 5;
  // wave handles 16 candidates; lane L and L+16 both work on candidate L&15
  const int c   = blockIdx.x * 128 + wid * 16 + (lane & 15);
  const int khi = (lane >= 16) ? 8 : 0;

  // Build 8-bit A-matrix (16x64): VGPR g, byte b holds flag for K per ISA layout
  v8i a = {};
#pragma unroll
  for (int g = 0; g < 8; ++g) {
    unsigned word = 0u;
#pragma unroll
    for (int b = 0; b < 4; ++b) {
      const int K = ((g >> 1) << 4) + ((g & 1) << 2) + b + khi;
      unsigned bit = 0u;
      if (K < TRAJ_LEN) {
        float2 p = traj[c * TRAJ_LEN + K];
        int col = (int)p.x + AREA_HALF;   // trunc toward zero, matches jnp.trunc
        int row = (int)p.y + AREA_HALF;
        int idx = row * AREA_SIZE + col;
        bit = (smask[idx >> 5] >> (idx & 31)) & 1u;
      }
      word |= bit << (8 * b);
    }
    a[g] = (int)word;
  }

  v8i bmat;
#pragma unroll
  for (int g = 0; g < 8; ++g) bmat[g] = 0x01010101;  // B = all ones (64x16 u8)
  v8i c0 = {};
  // D[m][n] = sum_k A[m][k] : off-DA count per candidate, broadcast over n
  v8i d = __builtin_amdgcn_wmma_i32_16x16x64_iu8(false, a, false, bmat, c0,
                                                 false, false);

  // Extract count[m] for lane m (m = 0..15):
  //  m<8 : VGPR m on lanes 0-15 ; m>=8 : VGPR m-8 on lanes 16-31
  int sel = d[lane & 7];
  int cnt = __shfl(sel, (lane < 8) ? lane : (lane + 8), 32);

  if (lane < 16)
    scores[c] = conf[c] - PENALTY * (float)cnt;
}

// ---------------- Kernel 2: top-6 (stable) + output gather -------------------
__global__ __launch_bounds__(256) void select_kernel(const float* __restrict__ traj,
                                                     const float* __restrict__ targ,
                                                     const float* __restrict__ scores,
                                                     float* __restrict__ out) {
  __shared__ float rv[256];
  __shared__ int   ri[256];
  __shared__ int   sSel[N_SELECT];
  __shared__ float sSelV[N_SELECT];
  const int tid = threadIdx.x;

  for (int it = 0; it < N_SELECT; ++it) {
    float bv = -3.402823466e38f;
    int   bi = 0x7fffffff;
    for (int i = tid; i < N_CAND; i += 256) {
      bool skip = false;
      for (int k = 0; k < it; ++k)
        if (sSel[k] == i) skip = true;
      if (skip) continue;
      float v = scores[i];
      if (v > bv || (v == bv && i < bi)) { bv = v; bi = i; }
    }
    rv[tid] = bv; ri[tid] = bi;
    __syncthreads();
    for (int s = 128; s > 0; s >>= 1) {
      if (tid < s) {
        float v2 = rv[tid + s]; int i2 = ri[tid + s];
        if (v2 > rv[tid] || (v2 == rv[tid] && i2 < ri[tid])) {
          rv[tid] = v2; ri[tid] = i2;
        }
      }
      __syncthreads();
    }
    if (tid == 0) { sSel[it] = ri[0]; sSelV[it] = rv[0]; }
    __syncthreads();
  }

  // out layout: trajectories[top] (720) | penalized[top] (6) | targets[:6] (720)
  const int TRAJ_F = TRAJ_LEN * 2;  // 120 floats per candidate
  for (int i = tid; i < N_SELECT * TRAJ_F; i += 256) {
    int s6 = sSel[i / TRAJ_F];
    out[i] = traj[s6 * TRAJ_F + (i % TRAJ_F)];
  }
  if (tid < N_SELECT) out[N_SELECT * TRAJ_F + tid] = sSelV[tid];
  for (int i = tid; i < N_SELECT * TRAJ_F; i += 256)
    out[N_SELECT * TRAJ_F + N_SELECT + i] = targ[i];
}

extern "C" void kernel_launch(void* const* d_in, const int* in_sizes, int n_in,
                              void* d_out, int out_size, void* d_ws, size_t ws_size,
                              hipStream_t stream) {
  const float* traj = (const float*)d_in[0];   // [262144,60,2] f32
  const float* conf = (const float*)d_in[1];   // [262144]      f32
  const float* targ = (const float*)d_in[2];   // [262144,60,2] f32
  const float* da   = (const float*)d_in[3];   // [224,224]     f32

  unsigned* maskw = (unsigned*)d_ws;                       // 1568 words (6272 B)
  float*    scores = (float*)((char*)d_ws + 8192);         // 262144 floats (1 MB)
  float*    out = (float*)d_out;

  pack_kernel<<<(MASK_WORDS + 255) / 256, 256, 0, stream>>>(da, maskw);
  score_kernel<<<N_CAND / 128, 256, 0, stream>>>((const float2*)traj, conf,
                                                 maskw, scores);
  select_kernel<<<1, 256, 0, stream>>>(traj, targ, scores, out);
}
